// ConditionalFeedForward_74122545594923
// MI455X (gfx1250) — matrix operand order repaired and checked by reading, not compiled, not run on hardware
//
#include <hip/hip_runtime.h>
#include <hip/hip_bf16.h>
#include <stdint.h>

#define T_TOK 4096
#define DIM   1024
#define NEXP  8
#define INTER 1408
#define TOPK  2
#define NROWS (T_TOK * TOPK)   // 8192 routed rows (sum of expert counts)

typedef __bf16 bf16;
typedef __attribute__((ext_vector_type(16))) __bf16 v16bf;
typedef __attribute__((ext_vector_type(8)))  __bf16 v8bf;
typedef __attribute__((ext_vector_type(8)))  float  v8f;
typedef __attribute__((ext_vector_type(4)))  int    v4i;

typedef __attribute__((address_space(1))) v4i gv4i;   // global int4
typedef __attribute__((address_space(3))) v4i lv4i;   // LDS int4

// ---------------- workspace layout ----------------
// [ints]   wsi[0..7] counts | wsi[8..15] base | wsi[16..23] cursor | wsi[32..8223] rowmap
// bytes 65536:            xb  [4096][1024]  bf16   ( 8,388,608 B)
// bytes 8,454,144:        wA  [8][1408][1024] bf16 (23,068,672 B)  w1 -> later reused for w2
// bytes 31,522,816:       wB  [8][1408][1024] bf16 (23,068,672 B)  w3
// bytes 54,591,488:       H   [8192][1408]  bf16   (23,068,672 B)
// total ~77.7 MB
#define WS_XB  ((size_t)65536)
#define WS_WA  (WS_XB + (size_t)8388608)
#define WS_WB  (WS_WA + (size_t)23068672)
#define WS_H   (WS_WB + (size_t)23068672)

// ---------------- async copy (gfx1250 ASYNCcnt path, guarded) ----------------
#if __has_builtin(__builtin_amdgcn_global_load_async_to_lds_b128)
#define ASYNC_COPY 1
#else
#define ASYNC_COPY 0
#endif

__device__ __forceinline__ void cp16(bf16* dstLds, const bf16* srcGlob) {
#if ASYNC_COPY
    __builtin_amdgcn_global_load_async_to_lds_b128((gv4i*)srcGlob, (lv4i*)dstLds, 0, 0);
#else
    *(v8bf*)dstLds = *(const v8bf*)srcGlob;
#endif
}

template <int N>
__device__ __forceinline__ void cp_wait() {
#if ASYNC_COPY
#if __has_builtin(__builtin_amdgcn_s_wait_asynccnt)
    __builtin_amdgcn_s_wait_asynccnt(N);
#else
    asm volatile("s_wait_asynccnt %0" :: "i"(N) : "memory");
#endif
#endif
}

// ---------------- routing kernels ----------------
__global__ void zero_counts_kernel(int* wsi) {
    if (threadIdx.x < 8) wsi[threadIdx.x] = 0;
}

__global__ void route_count_kernel(const long long* __restrict__ idx,
                                   int* __restrict__ wsi) {
    int i = blockIdx.x * 256 + threadIdx.x;
    if (i < NROWS) atomicAdd(&wsi[(int)idx[i]], 1);
}

__global__ void route_scan_kernel(int* wsi) {
    if (threadIdx.x == 0) {
        int acc = 0;
        for (int e = 0; e < NEXP; ++e) {
            wsi[8 + e]  = acc;
            wsi[16 + e] = acc;
            acc += wsi[e];
        }
    }
}

__global__ void route_scatter_kernel(const long long* __restrict__ idx,
                                     int* __restrict__ wsi) {
    int i = blockIdx.x * 256 + threadIdx.x;
    if (i < NROWS) {
        int e   = (int)idx[i];
        int pos = atomicAdd(&wsi[16 + e], 1);
        wsi[32 + pos] = i;
    }
}

// ---------------- fp32 -> bf16 bulk convert (8 elems / thread) ----------------
__device__ __forceinline__ v8bf cvt8(const float4 a, const float4 b) {
    v8bf v;
    v[0] = (__bf16)a.x; v[1] = (__bf16)a.y; v[2] = (__bf16)a.z; v[3] = (__bf16)a.w;
    v[4] = (__bf16)b.x; v[5] = (__bf16)b.y; v[6] = (__bf16)b.z; v[7] = (__bf16)b.w;
    return v;
}

__global__ __launch_bounds__(256) void f32_to_bf16_kernel(const float* __restrict__ src,
                                                          bf16* __restrict__ dst, int n8) {
    int i = blockIdx.x * 256 + threadIdx.x;
    if (i < n8) {
        const float4* s = (const float4*)(src + (size_t)i * 8);
        float4 a = s[0], b = s[1];
        *(v8bf*)(dst + (size_t)i * 8) = cvt8(a, b);
    }
}

// ---------------- WMMA fragment helpers ----------------
// Pointers passed here must be derived directly from a __shared__ array so the
// compiler keeps them in addrspace(3) and emits ds_load_b128 (NOT flat_load).
union FragU { v16bf v; struct { v8bf lo, hi; } h; };

// A 16x32 bf16: lane%16 = M row; half-wave splits K in 8-chunks within each 16
__device__ __forceinline__ v16bf frag_A(const bf16* tile, int rowOff, int half) {
    const bf16* rowp = tile + rowOff;
    FragU f;
    f.h.lo = *(const v8bf*)(rowp + half * 8);
    f.h.hi = *(const v8bf*)(rowp + 16 + half * 8);
    return f.v;
}
// B 32x16 bf16: lane%16 = N col; half-wave splits K in 8-chunks within each 16
__device__ __forceinline__ v16bf frag_B(const bf16* tile, int rowOff, int half) {
    const bf16* rowp = tile + rowOff;
    FragU f;
    f.h.lo = *(const v8bf*)(rowp + half * 8);
    f.h.hi = *(const v8bf*)(rowp + 16 + half * 8);
    return f.v;
}

#define WMMA_BF16(A, B, C) \
    __builtin_amdgcn_wmma_f32_16x16x32_bf16(false, (A), false, (B), (short)0, (C), false, false)

// ---------------- GEMM1: H = silu(X*W1^T) * (X*W3^T), grouped by expert ----------------
// block tile 64(M) x 128(N), K-step 32, double-buffered LDS stages (5 async b128/thread)
// 8 waves in 2(M) x 4(N); wave tile 32x32 -> 8 WMMA / K-step
// grid: (INTER/128, NROWS/64, NEXP)
__global__ __launch_bounds__(256) void moe_gemm1_kernel(
    const bf16* __restrict__ xb, const bf16* __restrict__ w1b,
    const bf16* __restrict__ w3b, const int* __restrict__ wsi,
    bf16* __restrict__ H)
{
    const int e     = blockIdx.z;
    const int cnt   = wsi[e];
    const int tileM = blockIdx.y;
    if (tileM * 64 >= cnt) return;                  // uniform exit: EXEC stays full
    const int baseR = wsi[8 + e] + tileM * 64;
    const int endR  = wsi[8 + e] + cnt;
    const int nBase = blockIdx.x * 128;
    const int* rowmap = wsi + 32;

    __shared__ __align__(16) bf16 sX [2][64  * 40];
    __shared__ __align__(16) bf16 sW1[2][128 * 40];
    __shared__ __align__(16) bf16 sW3[2][128 * 40];
    __shared__ int sTok[64];

    const int tid = threadIdx.x;
    if (tid < 64) {
        int r  = baseR + tid;
        int ft = (r < endR) ? rowmap[min(r, NROWS - 1)] : 0;
        sTok[tid] = ft >> 1;
    }
    __syncthreads();

    const int xr  = tid >> 2;            // 0..63
    const int xc  = (tid & 3) * 8;       // 0,8,16,24
    const int wr0 = tid >> 2;            // 0..63
    const int wr1 = wr0 + 64;            // 64..127
    const int tok = sTok[xr];

    const bf16* xrow = xb  + (size_t)tok * DIM + xc;
    const bf16* w1e  = w1b + (size_t)e * INTER * DIM;
    const bf16* w3e  = w3b + (size_t)e * INTER * DIM;
    const bf16* w1r0 = w1e + (size_t)(nBase + wr0) * DIM + xc;
    const bf16* w1r1 = w1e + (size_t)(nBase + wr1) * DIM + xc;
    const bf16* w3r0 = w3e + (size_t)(nBase + wr0) * DIM + xc;
    const bf16* w3r1 = w3e + (size_t)(nBase + wr1) * DIM + xc;

    const int lane = tid & 31;
    const int half = lane >> 4;
    const int l16  = lane & 15;
    const int w    = tid >> 5;
    const int wm   = (w & 1) * 32;       // wave M offset
    const int wn   = (w >> 1) * 32;      // wave N offset

    // per-lane row offsets within a tile (buffer chosen by direct shared indexing)
    const int offA0 = (wm      + l16) * 40;
    const int offA1 = (wm + 16 + l16) * 40;
    const int offB0 = (wn      + l16) * 40;
    const int offB1 = (wn + 16 + l16) * 40;

    auto issue = [&](int k0, int buf) {
        cp16(&sX [buf][xr  * 40 + xc], xrow + k0);
        cp16(&sW1[buf][wr0 * 40 + xc], w1r0 + k0);
        cp16(&sW1[buf][wr1 * 40 + xc], w1r1 + k0);
        cp16(&sW3[buf][wr0 * 40 + xc], w3r0 + k0);
        cp16(&sW3[buf][wr1 * 40 + xc], w3r1 + k0);
    };

    v8f c1_00 = {}, c1_01 = {}, c1_10 = {}, c1_11 = {};
    v8f c3_00 = {}, c3_01 = {}, c3_10 = {}, c3_11 = {};

    issue(0, 0);                                    // prologue: stage 0
    const int KSTEPS = DIM / 32;                    // 32
#pragma unroll 1
    for (int i = 0; i < KSTEPS; ++i) {
        const int buf = i & 1;
        if (i + 1 < KSTEPS) { issue((i + 1) * 32, buf ^ 1); cp_wait<5>(); }
        else                { cp_wait<0>(); }
        __syncthreads();                            // stage i resident in LDS for all waves

        // load all fragments first (ds_load_b128), then back-to-back WMMAs
        v16bf A0  = frag_A(&sX [buf][0], offA0, half);
        v16bf A1  = frag_A(&sX [buf][0], offA1, half);
        v16bf B10 = frag_B(&sW1[buf][0], offB0, half);
        v16bf B11 = frag_B(&sW1[buf][0], offB1, half);
        v16bf B30 = frag_B(&sW3[buf][0], offB0, half);
        v16bf B31 = frag_B(&sW3[buf][0], offB1, half);

        c1_00 = WMMA_BF16(A0, B10, c1_00);
        c1_10 = WMMA_BF16(A1, B10, c1_10);
        c1_01 = WMMA_BF16(A0, B11, c1_01);
        c1_11 = WMMA_BF16(A1, B11, c1_11);
        c3_00 = WMMA_BF16(A0, B30, c3_00);
        c3_10 = WMMA_BF16(A1, B30, c3_10);
        c3_01 = WMMA_BF16(A0, B31, c3_01);
        c3_11 = WMMA_BF16(A1, B31, c3_11);

        __syncthreads();                            // release buf for stage i+2 writers
    }

    // epilogue: h = silu(x1) * x3 -> H (bf16); one row pointer per subtile, stride INTER
    const int col = l16;
    for (int i = 0; i < 2; ++i) {
        for (int j = 0; j < 2; ++j) {
            v8f c1 = i ? (j ? c1_11 : c1_10) : (j ? c1_01 : c1_00);
            v8f c3 = i ? (j ? c3_11 : c3_10) : (j ? c3_01 : c3_00);
            int n  = nBase + wn + 16 * j + col;
            int r0 = baseR + wm + 16 * i + half * 8;
            bf16* p = H + (size_t)r0 * INTER + n;
            for (int v = 0; v < 8; ++v) {
                if (r0 + v < endR) {
                    float g = c1[v];
                    float s = g / (1.0f + __expf(-g));
                    *p = (bf16)(s * c3[v]);
                }
                p += INTER;
            }
        }
    }
}

// ---------------- GEMM2: out = H * W2^T, scattered to [T, TOPK, DIM] ----------------
// block tile 64(M) x 128(N), double-buffered (3 async b128/thread per stage)
// grid: (DIM/128, NROWS/64, NEXP)
__global__ __launch_bounds__(256) void moe_gemm2_kernel(
    const bf16* __restrict__ H, const bf16* __restrict__ w2b,
    const int* __restrict__ wsi, float* __restrict__ out)
{
    const int e     = blockIdx.z;
    const int cnt   = wsi[e];
    const int tileM = blockIdx.y;
    if (tileM * 64 >= cnt) return;
    const int baseR = wsi[8 + e] + tileM * 64;
    const int endR  = wsi[8 + e] + cnt;
    const int nBase = blockIdx.x * 128;
    const int* rowmap = wsi + 32;

    __shared__ __align__(16) bf16 sH[2][64  * 40];
    __shared__ __align__(16) bf16 sW[2][128 * 40];
    __shared__ int sFT[64];

    const int tid = threadIdx.x;
    if (tid < 64) {
        int r = baseR + tid;
        sFT[tid] = (r < endR) ? rowmap[min(r, NROWS - 1)] : 0;
    }
    __syncthreads();

    const int xr  = tid >> 2;
    const int xc  = (tid & 3) * 8;
    const int wr0 = tid >> 2;
    const int wr1 = wr0 + 64;
    const int hRow = min(baseR + xr, NROWS - 1);    // invalid rows masked at store

    const bf16* hrow = H + (size_t)hRow * INTER + xc;
    const bf16* w2e  = w2b + (size_t)e * DIM * INTER;
    const bf16* w2r0 = w2e + (size_t)(nBase + wr0) * INTER + xc;
    const bf16* w2r1 = w2e + (size_t)(nBase + wr1) * INTER + xc;

    const int lane = tid & 31;
    const int half = lane >> 4;
    const int l16  = lane & 15;
    const int w    = tid >> 5;
    const int wm   = (w & 1) * 32;
    const int wn   = (w >> 1) * 32;

    const int offA0 = (wm      + l16) * 40;
    const int offA1 = (wm + 16 + l16) * 40;
    const int offB0 = (wn      + l16) * 40;
    const int offB1 = (wn + 16 + l16) * 40;

    auto issue = [&](int k0, int buf) {
        cp16(&sH[buf][xr  * 40 + xc], hrow + k0);
        cp16(&sW[buf][wr0 * 40 + xc], w2r0 + k0);
        cp16(&sW[buf][wr1 * 40 + xc], w2r1 + k0);
    };

    v8f c00 = {}, c01 = {}, c10 = {}, c11 = {};

    issue(0, 0);
    const int KSTEPS = INTER / 32;                  // 44
#pragma unroll 1
    for (int i = 0; i < KSTEPS; ++i) {
        const int buf = i & 1;
        if (i + 1 < KSTEPS) { issue((i + 1) * 32, buf ^ 1); cp_wait<3>(); }
        else                { cp_wait<0>(); }
        __syncthreads();

        v16bf A0 = frag_A(&sH[buf][0], offA0, half);
        v16bf A1 = frag_A(&sH[buf][0], offA1, half);
        v16bf B0 = frag_B(&sW[buf][0], offB0, half);
        v16bf B1 = frag_B(&sW[buf][0], offB1, half);

        c00 = WMMA_BF16(A0, B0, c00);
        c10 = WMMA_BF16(A1, B0, c10);
        c01 = WMMA_BF16(A0, B1, c01);
        c11 = WMMA_BF16(A1, B1, c11);

        __syncthreads();
    }

    const int col = l16;
    for (int i = 0; i < 2; ++i) {
        for (int j = 0; j < 2; ++j) {
            v8f c = i ? (j ? c11 : c10) : (j ? c01 : c00);
            int n  = nBase + wn + 16 * j + col;
            int mB = wm + 16 * i + half * 8;
            for (int v = 0; v < 8; ++v) {
                int r = baseR + mB + v;
                if (r < endR) {
                    int ft = sFT[r - baseR];
                    out[(size_t)ft * DIM + n] = c[v];
                }
            }
        }
    }
}

// ---------------- launch ----------------
extern "C" void kernel_launch(void* const* d_in, const int* in_sizes, int n_in,
                              void* d_out, int out_size, void* d_ws, size_t ws_size,
                              hipStream_t stream) {
    const float*     x   = (const float*)d_in[0];
    const long long* idx = (const long long*)d_in[1];   // int64 expert indices
    const float*     w1  = (const float*)d_in[2];
    const float*     w2  = (const float*)d_in[3];
    const float*     w3  = (const float*)d_in[4];
    float* out = (float*)d_out;

    int*  wsi = (int*)d_ws;
    bf16* xb  = (bf16*)((char*)d_ws + WS_XB);
    bf16* wA  = (bf16*)((char*)d_ws + WS_WA);   // w1 bf16, later overwritten with w2 bf16
    bf16* wB  = (bf16*)((char*)d_ws + WS_WB);   // w3 bf16
    bf16* H   = (bf16*)((char*)d_ws + WS_H);

    // routing
    zero_counts_kernel  <<<1, 32, 0, stream>>>(wsi);
    route_count_kernel  <<<NROWS / 256, 256, 0, stream>>>(idx, wsi);
    route_scan_kernel   <<<1, 32, 0, stream>>>(wsi);
    route_scatter_kernel<<<NROWS / 256, 256, 0, stream>>>(idx, wsi);

    // one-shot fp32 -> bf16 conversions
    const int n8x = T_TOK * DIM / 8;           // 524288
    const int n8w = NEXP * INTER * DIM / 8;    // 1441792
    f32_to_bf16_kernel<<<n8x / 256, 256, 0, stream>>>(x,  xb, n8x);
    f32_to_bf16_kernel<<<n8w / 256, 256, 0, stream>>>(w1, wA, n8w);
    f32_to_bf16_kernel<<<n8w / 256, 256, 0, stream>>>(w3, wB, n8w);

    dim3 g1(INTER / 128, NROWS / 64, NEXP);    // 11 x 128 x 8 (most tiles exit early)
    moe_gemm1_kernel<<<g1, 256, 0, stream>>>(xb, wA, wB, wsi, H);

    // reuse wA for w2 (same element count: E*DIM*INTER)
    f32_to_bf16_kernel<<<n8w / 256, 256, 0, stream>>>(w2, wA, n8w);

    dim3 g2(DIM / 128, NROWS / 64, NEXP);      // 8 x 128 x 8
    moe_gemm2_kernel<<<g2, 256, 0, stream>>>(H, wA, wsi, out);
}
